// ProtoNet_71219147702272
// MI455X (gfx1250) — matrix operand order repaired and checked by reading, not compiled, not run on hardware
//
#include <hip/hip_runtime.h>
#include <hip/hip_bf16.h>

// ---------------------------------------------------------------------------
// ProtoNet LeNet forward for MI455X (gfx1250, wave32, WMMA).
//   conv1+relu+pool (f32 VALU)            -> ws1 [N,6,14,14] f32
//   conv2+relu+pool (f32 acc, f16 store)  -> ws2 [N,416] f16 (zero-padded)
//   prep_weights: fc weights -> padded f16 (done once per launch, tiny)
//   fc1/fc2/fc3 via v_wmma_f32_16x16x32_f16, f16 operands staged in LDS,
//   f32 accumulate                         -> emb [N,64] f32
//   softmax + |s - proto| head             -> d_out
// ---------------------------------------------------------------------------

typedef __attribute__((ext_vector_type(16))) _Float16 v16h;
typedef __attribute__((ext_vector_type(8)))  _Float16 v8h;
typedef __attribute__((ext_vector_type(8)))  float    v8f;

#define K1PAD 416   // fc1 K: 400 -> 416 (13 x 32)
#define N1PAD 128   // fc1 N: 120 -> 128
#define K2PAD 128   // fc2 K: 120 -> 128 (4 x 32)
#define N2PAD 96    // fc2 N: 84  -> 96
#define K3PAD 96    // fc3 K: 84  -> 96  (3 x 32)
#define N3PAD 64    // fc3 N: 64

// ----------------------------- conv1 + pool --------------------------------
__global__ __launch_bounds__(256)
void conv1_pool(const float* __restrict__ x, const float* __restrict__ w,
                const float* __restrict__ b, float* __restrict__ out, int N) {
    int idx = blockIdx.x * blockDim.x + threadIdx.x;
    int total = N * 6 * 14 * 14;
    if (idx >= total) return;
    int j = idx % 14;
    int t = idx / 14;
    int i = t % 14;  t /= 14;
    int c = t % 6;
    int n = t / 6;

    const float* xi = x + (size_t)n * 784;
    float wv[25];
#pragma unroll
    for (int u = 0; u < 25; ++u) wv[u] = w[c * 25 + u];
    float bias = b[c];

    float best = 0.0f;   // relu output >= 0
#pragma unroll
    for (int dy = 0; dy < 2; ++dy) {
#pragma unroll
        for (int dx = 0; dx < 2; ++dx) {
            int ci = 2 * i + dy;
            int cj = 2 * j + dx;
            float s = bias;
#pragma unroll
            for (int u = 0; u < 5; ++u) {
                int yy = ci + u - 2;
                if ((unsigned)yy >= 28u) continue;
#pragma unroll
                for (int v = 0; v < 5; ++v) {
                    int xx = cj + v - 2;
                    if ((unsigned)xx >= 28u) continue;
                    s += xi[yy * 28 + xx] * wv[u * 5 + v];
                }
            }
            best = fmaxf(best, fmaxf(s, 0.0f));
        }
    }
    out[idx] = best;
}

// ------------------- conv2 + pool (writes padded f16) ----------------------
__global__ __launch_bounds__(256)
void conv2_pool(const float* __restrict__ in, const float* __restrict__ w,
                const float* __restrict__ b, _Float16* __restrict__ out, int N) {
    int idx = blockIdx.x * blockDim.x + threadIdx.x;
    int total = N * 16 * 5 * 5;
    if (idx >= total) return;
    int j = idx % 5;
    int t = idx / 5;
    int i = t % 5;   t /= 5;
    int c = t % 16;
    int n = t / 16;

    const float* xi = in + (size_t)n * 1176;   // [6][14][14]
    float bias = b[c];

    float best = 0.0f;
#pragma unroll
    for (int dy = 0; dy < 2; ++dy) {
#pragma unroll
        for (int dx = 0; dx < 2; ++dx) {
            int ci = 2 * i + dy;               // valid conv, 0..9
            int cj = 2 * j + dx;
            float s = bias;
            for (int ic = 0; ic < 6; ++ic) {
                const float* ip = xi + ic * 196;
                const float* wp = w + c * 150 + ic * 25;
#pragma unroll
                for (int u = 0; u < 5; ++u) {
#pragma unroll
                    for (int v = 0; v < 5; ++v) {
                        s += ip[(ci + u) * 14 + (cj + v)] * wp[u * 5 + v];
                    }
                }
            }
            best = fmaxf(best, fmaxf(s, 0.0f));
        }
    }
    out[(size_t)n * K1PAD + c * 25 + i * 5 + j] = (_Float16)best;
}

// zero the padding columns 400..415 of ws2
__global__ __launch_bounds__(256)
void pad_ws2(_Float16* __restrict__ out, int N) {
    int idx = blockIdx.x * blockDim.x + threadIdx.x;
    if (idx >= N * 16) return;
    int n = idx / 16;
    int k = 400 + (idx % 16);
    out[(size_t)n * K1PAD + k] = (_Float16)0.0f;
}

// ----------- one-time conversion of FC weights to padded f16 ---------------
__global__ __launch_bounds__(256)
void prep_weights(const float* __restrict__ f1w, const float* __restrict__ f2w,
                  const float* __restrict__ f3w,
                  _Float16* __restrict__ w1, _Float16* __restrict__ w2,
                  _Float16* __restrict__ w3) {
    const int n1 = N1PAD * K1PAD;   // 53248
    const int n2 = N2PAD * K2PAD;   // 12288
    const int n3 = N3PAD * K3PAD;   // 6144
    int idx = blockIdx.x * blockDim.x + threadIdx.x;
    if (idx < n1) {
        int r = idx / K1PAD, k = idx % K1PAD;
        w1[idx] = (_Float16)((r < 120 && k < 400) ? f1w[r * 400 + k] : 0.0f);
    } else if (idx < n1 + n2) {
        int t = idx - n1;
        int r = t / K2PAD, k = t % K2PAD;
        w2[t] = (_Float16)((r < 84 && k < 120) ? f2w[r * 120 + k] : 0.0f);
    } else if (idx < n1 + n2 + n3) {
        int t = idx - n1 - n2;
        int r = t / K3PAD, k = t % K3PAD;
        w3[t] = (_Float16)((k < 84) ? f3w[r * 84 + k] : 0.0f);
    }
}

// --------------------------- WMMA fragment loads ---------------------------
// A fragment (16x32 f16, ISA 7.12.2): lane = kh*16 + m.
//   VGPR 0..3: K = kh*8 + {0..7};  VGPR 4..7: K = 16 + kh*8 + {0..7}.
// Two contiguous 16-byte runs per lane -> two b128 loads, no conversion.
__device__ __forceinline__ v16h frag_a16(const _Float16* base, int ld, int m,
                                         int kbase, int kh) {
    v8h lo = *(const v8h*)(base + (size_t)m * ld + kbase + kh * 8);
    v8h hi = *(const v8h*)(base + (size_t)m * ld + kbase + 16 + kh * 8);
    return __builtin_shufflevector(lo, hi, 0, 1, 2, 3, 4, 5, 6, 7,
                                   8, 9, 10, 11, 12, 13, 14, 15);
}

// B fragment (32x16 f16): lane = kh*16 + n; same K order as A:
//   VGPR 0..3: K = kh*8 + {0..7};  VGPR 4..7: K = 16 + kh*8 + {0..7} of weight row n.
__device__ __forceinline__ v16h frag_b16(const _Float16* __restrict__ W, int ldw,
                                         int n, int kbase, int kh) {
    v8h lo = *(const v8h*)(W + (size_t)n * ldw + kbase + kh * 8);
    v8h hi = *(const v8h*)(W + (size_t)n * ldw + kbase + 16 + kh * 8);
    return __builtin_shufflevector(lo, hi, 0, 1, 2, 3, 4, 5, 6, 7,
                                   8, 9, 10, 11, 12, 13, 14, 15);
}

// ------------------------------- FC stack ----------------------------------
// 128 threads = 4 wave32; each wave owns a 16-image tile.
// K-chunk-outer loops: A fragment loaded once per chunk, all N-tiles live.
__global__ __launch_bounds__(128)
void fc_stack(const _Float16* __restrict__ xin,                    // [N][416]
              const _Float16* __restrict__ w1,                     // [128][416]
              const _Float16* __restrict__ w2,                     // [96][128]
              const _Float16* __restrict__ w3,                     // [64][96]
              const float* __restrict__ f1b, const float* __restrict__ f2b,
              const float* __restrict__ f3b,
              float* __restrict__ emb, int N) {
    __shared__ __align__(16) _Float16 H[4][16][K2PAD];   // fc1 out (f16)
    __shared__ __align__(16) _Float16 G[4][16][K3PAD];   // fc2 out (f16)

    int wave = threadIdx.x >> 5;
    int lane = threadIdx.x & 31;
    int kh   = lane >> 4;
    int l16  = lane & 15;
    int imgbase = (blockIdx.x * 4 + wave) * 16;
    if (imgbase >= N) return;

    const _Float16* A0 = xin + (size_t)imgbase * K1PAD;

    // ---- fc1: [16,416] x [416,128] ; 13 K-chunks, 8 N-tiles ----
    {
        v8f acc[8];
#pragma unroll
        for (int t = 0; t < 8; ++t) acc[t] = (v8f){};
#pragma unroll
        for (int kc = 0; kc < 13; ++kc) {
            v16h a = frag_a16(A0, K1PAD, l16, kc * 32, kh);
#pragma unroll
            for (int t = 0; t < 8; ++t) {
                v16h b = frag_b16(w1, K1PAD, t * 16 + l16, kc * 32, kh);
                acc[t] = __builtin_amdgcn_wmma_f32_16x16x32_f16(
                    false, a, false, b, (short)0, acc[t], false, false);
            }
        }
#pragma unroll
        for (int t = 0; t < 8; ++t) {
#pragma unroll
            for (int r = 0; r < 8; ++r) {
                int row = r + 8 * kh;
                int col = t * 16 + l16;
                float v = (col < 120) ? fmaxf(acc[t][r] + f1b[col], 0.0f) : 0.0f;
                H[wave][row][col] = (_Float16)v;
            }
        }
    }

    // ---- fc2: [16,128] x [128,96] ; 4 K-chunks, 6 N-tiles ----
    {
        v8f acc[6];
#pragma unroll
        for (int t = 0; t < 6; ++t) acc[t] = (v8f){};
#pragma unroll
        for (int kc = 0; kc < 4; ++kc) {
            v16h a = frag_a16(&H[wave][0][0], K2PAD, l16, kc * 32, kh);
#pragma unroll
            for (int t = 0; t < 6; ++t) {
                v16h b = frag_b16(w2, K2PAD, t * 16 + l16, kc * 32, kh);
                acc[t] = __builtin_amdgcn_wmma_f32_16x16x32_f16(
                    false, a, false, b, (short)0, acc[t], false, false);
            }
        }
#pragma unroll
        for (int t = 0; t < 6; ++t) {
#pragma unroll
            for (int r = 0; r < 8; ++r) {
                int row = r + 8 * kh;
                int col = t * 16 + l16;
                float v = (col < 84) ? fmaxf(acc[t][r] + f2b[col], 0.0f) : 0.0f;
                G[wave][row][col] = (_Float16)v;
            }
        }
    }

    // ---- fc3: [16,96] x [96,64] ; 3 K-chunks, 4 N-tiles ----
    {
        v8f acc[4];
#pragma unroll
        for (int t = 0; t < 4; ++t) acc[t] = (v8f){};
#pragma unroll
        for (int kc = 0; kc < 3; ++kc) {
            v16h a = frag_a16(&G[wave][0][0], K3PAD, l16, kc * 32, kh);
#pragma unroll
            for (int t = 0; t < 4; ++t) {
                v16h b = frag_b16(w3, K3PAD, t * 16 + l16, kc * 32, kh);
                acc[t] = __builtin_amdgcn_wmma_f32_16x16x32_f16(
                    false, a, false, b, (short)0, acc[t], false, false);
            }
        }
#pragma unroll
        for (int t = 0; t < 4; ++t) {
#pragma unroll
            for (int r = 0; r < 8; ++r) {
                int row = r + 8 * kh;
                int col = t * 16 + l16;
                if (imgbase + row < N)
                    emb[(size_t)(imgbase + row) * 64 + col] = acc[t][r] + f3b[col];
            }
        }
    }
}

// ------------------------- softmax + |s - proto| ---------------------------
__global__ __launch_bounds__(256)
void head(const float* __restrict__ emb, float* __restrict__ out, int N) {
    int row  = blockIdx.x * 8 + (threadIdx.x >> 5);
    int lane = threadIdx.x & 31;
    if (row >= N) return;

    float v0 = emb[(size_t)row * 64 + lane];
    float v1 = emb[(size_t)row * 64 + 32 + lane];

    float m = fmaxf(v0, v1);
#pragma unroll
    for (int o = 16; o >= 1; o >>= 1) m = fmaxf(m, __shfl_xor(m, o, 32));

    float e0 = expf(v0 - m);
    float e1 = expf(v1 - m);
    float s = e0 + e1;
#pragma unroll
    for (int o = 16; o >= 1; o >>= 1) s += __shfl_xor(s, o, 32);
    float inv = 1.0f / s;

    float p0 = emb[lane] * 0.2f;        // proto = emb[0] / 5
    float p1 = emb[32 + lane] * 0.2f;

    out[(size_t)row * 64 + lane]      = fabsf(e0 * inv - p0);
    out[(size_t)row * 64 + 32 + lane] = fabsf(e1 * inv - p1);
}

// ------------------------------- launcher ----------------------------------
extern "C" void kernel_launch(void* const* d_in, const int* in_sizes, int n_in,
                              void* d_out, int out_size, void* d_ws, size_t ws_size,
                              hipStream_t stream) {
    const float* x   = (const float*)d_in[0];
    const float* c1w = (const float*)d_in[1];
    const float* c1b = (const float*)d_in[2];
    const float* c2w = (const float*)d_in[3];
    const float* c2b = (const float*)d_in[4];
    const float* f1w = (const float*)d_in[5];
    const float* f1b = (const float*)d_in[6];
    const float* f2w = (const float*)d_in[7];
    const float* f2b = (const float*)d_in[8];
    const float* f3w = (const float*)d_in[9];
    const float* f3b = (const float*)d_in[10];

    int N = in_sizes[0] / 784;                 // 20480

    // workspace layout (f32 region first, then f16 region; all 16B aligned)
    float* ws1 = (float*)d_ws;                         // [N][1176] f32
    float* emb = ws1 + (size_t)N * 1176;               // [N][64]   f32
    _Float16* ws2 = (_Float16*)(emb + (size_t)N * 64); // [N][416]  f16
    _Float16* w1  = ws2 + (size_t)N * K1PAD;           // [128][416]
    _Float16* w2  = w1 + N1PAD * K1PAD;                // [96][128]
    _Float16* w3  = w2 + N2PAD * K2PAD;                // [64][96]

    {
        int total = N1PAD * K1PAD + N2PAD * K2PAD + N3PAD * K3PAD;
        prep_weights<<<(total + 255) / 256, 256, 0, stream>>>(f1w, f2w, f3w,
                                                              w1, w2, w3);
    }
    {
        int total = N * 6 * 14 * 14;
        conv1_pool<<<(total + 255) / 256, 256, 0, stream>>>(x, c1w, c1b, ws1, N);
    }
    {
        int total = N * 16 * 5 * 5;
        conv2_pool<<<(total + 255) / 256, 256, 0, stream>>>(ws1, c2w, c2b, ws2, N);
        pad_ws2<<<(N * 16 + 255) / 256, 256, 0, stream>>>(ws2, N);
    }
    {
        int blocks = (N + 63) / 64;            // 4 waves x 16 images per block
        fc_stack<<<blocks, 128, 0, stream>>>(ws2, w1, w2, w3,
                                             f1b, f2b, f3b, emb, N);
    }
    {
        int blocks = (N + 7) / 8;
        head<<<blocks, 256, 0, stream>>>(emb, (float*)d_out, N);
    }
}
